// HSGSLanguageModel_86079734547003
// MI455X (gfx1250) — matrix operand order repaired and checked by reading, not compiled, not run on hardware
//
#include <hip/hip_runtime.h>

#define Vv   32000
#define DSd  64
#define DFd  256
#define Hh   8
#define Bb   2
#define Ll   1024
#define NBn  512
#define NROW (Bb*Ll)   // 2048

typedef __attribute__((ext_vector_type(16))) _Float16 v16h;
typedef __attribute__((ext_vector_type(8)))  _Float16 v8h;
typedef __attribute__((ext_vector_type(4)))  _Float16 v4h;
typedef __attribute__((ext_vector_type(8)))  float    v8f;
typedef __attribute__((ext_vector_type(4)))  float    v4f;

__device__ __forceinline__ float sigm(float x){ return 1.0f/(1.0f+__expf(-x)); }

// ---- WMMA fragment loads straight from row-major global f16 ----------------
// A (16x32, f16): lane(0-15)=M, k-halves: idx0-7 -> k=8*laneHi+0..7,
//                 idx8-15 -> k=16+8*laneHi+0..7  (two 16B chunks at p, p+16)
__device__ __forceinline__ v16h load_a_frag(const _Float16* p){
  v8h lo = *(const v8h*)p;
  v8h hi = *(const v8h*)(p + 16);
  v16h r;
#pragma unroll
  for (int i = 0; i < 8; ++i){ r[i] = lo[i]; r[i+8] = hi[i]; }
  return r;
}
// B (32x16, f16) from (N,K) row-major memory: lane&15=N, idx0-7 -> k=8*laneHi+0..7,
// idx8-15 -> k=16+8*laneHi+0..7  (two 16B chunks at p, p+16)
__device__ __forceinline__ v16h load_b_frag(const _Float16* p){
  v8h lo = *(const v8h*)p;
  v8h hi = *(const v8h*)(p + 16);
  v16h r;
#pragma unroll
  for (int i = 0; i < 8; ++i){ r[i] = lo[i]; r[i+8] = hi[i]; }
  return r;
}

__device__ __forceinline__ v8f wmma16(v16h a, v16h b, v8f c){
  return __builtin_amdgcn_wmma_f32_16x16x32_f16(false, a, false, b,
                                                (short)0, c, false, false);
}

// ---- Generic C = A @ B^T (+bias, +acc) ; A:(M,K) f16, B:(N,K) f16 ----------
// Block = 256 thr (8 waves): 256(M) x 64(N) block tile; each wave 32x64
// (2 M-subtiles x 4 N-subtiles = 8 wmma/k-step, 12 b128 loads/k-step),
// software-pipelined so next k-step loads overlap current WMMAs.
// Batched via grid.z: Aoff=z*aStride, Boff=(z/bDiv)*bStride,
// Coff=(z/cDiv)*cOuter+(z%cDiv)*cInner.  M%256==0, N%64==0, K%32==0 assumed.
template <typename CT>
__global__ __launch_bounds__(256)
void gemm_tn(const _Float16* __restrict__ A, const _Float16* __restrict__ Bm,
             const float* __restrict__ bias, const float* __restrict__ Cacc,
             CT* __restrict__ C, int M, int N, int K, int ldc,
             long aStride, long bStride, int bDiv,
             long cOuter, long cInner, int cDiv)
{
  const int z   = blockIdx.z;
  const _Float16* Az = A  + (long)z * aStride;
  const _Float16* Bz = Bm + (long)(z / bDiv) * bStride;
  const long cOff = (long)(z / cDiv) * cOuter + (long)(z % cDiv) * cInner;

  const int lane = threadIdx.x & 31;
  const int wv   = threadIdx.x >> 5;
  const int m0   = blockIdx.y * 256 + wv * 32;
  const int n0   = blockIdx.x * 64;
  if (m0 >= M || n0 >= N) return;
  const int lN = lane & 15, lH = lane >> 4;

  v8f acc[2][4] = {};
  const _Float16* aRow0 = Az + (long)(m0 + lN) * K + lH * 8;
  const _Float16* aRow1 = aRow0 + (long)16 * K;
  const _Float16* bRow  = Bz + (long)(n0 + lN) * K + lH * 8;
  const long bT = (long)16 * K;

  v16h a0 = load_a_frag(aRow0);
  v16h a1 = load_a_frag(aRow1);
  v16h b0 = load_b_frag(bRow);
  v16h b1 = load_b_frag(bRow + bT);
  v16h b2 = load_b_frag(bRow + 2 * bT);
  v16h b3 = load_b_frag(bRow + 3 * bT);

  for (int k0 = 32; k0 < K; k0 += 32){
    if (k0 + 32 < K) __builtin_prefetch((const void*)(aRow0 + k0 + 32), 0, 3);
    v16h a0n = load_a_frag(aRow0 + k0);
    v16h a1n = load_a_frag(aRow1 + k0);
    v16h b0n = load_b_frag(bRow + k0);
    v16h b1n = load_b_frag(bRow + bT + k0);
    v16h b2n = load_b_frag(bRow + 2 * bT + k0);
    v16h b3n = load_b_frag(bRow + 3 * bT + k0);
    acc[0][0] = wmma16(a0, b0, acc[0][0]);
    acc[0][1] = wmma16(a0, b1, acc[0][1]);
    acc[0][2] = wmma16(a0, b2, acc[0][2]);
    acc[0][3] = wmma16(a0, b3, acc[0][3]);
    acc[1][0] = wmma16(a1, b0, acc[1][0]);
    acc[1][1] = wmma16(a1, b1, acc[1][1]);
    acc[1][2] = wmma16(a1, b2, acc[1][2]);
    acc[1][3] = wmma16(a1, b3, acc[1][3]);
    a0 = a0n; a1 = a1n; b0 = b0n; b1 = b1n; b2 = b2n; b3 = b3n;
  }
  acc[0][0] = wmma16(a0, b0, acc[0][0]);
  acc[0][1] = wmma16(a0, b1, acc[0][1]);
  acc[0][2] = wmma16(a0, b2, acc[0][2]);
  acc[0][3] = wmma16(a0, b3, acc[0][3]);
  acc[1][0] = wmma16(a1, b0, acc[1][0]);
  acc[1][1] = wmma16(a1, b1, acc[1][1]);
  acc[1][2] = wmma16(a1, b2, acc[1][2]);
  acc[1][3] = wmma16(a1, b3, acc[1][3]);

  // C/D layout: lanes0-15 N=lane,M=r ; lanes16-31 N=lane-16,M=r+8
#pragma unroll
  for (int s = 0; s < 2; ++s){
#pragma unroll
    for (int t = 0; t < 4; ++t){
#pragma unroll
      for (int r = 0; r < 8; ++r){
        int m = m0 + s * 16 + r + 8 * lH;
        int n = n0 + t * 16 + lN;
        long ci = cOff + (long)m * ldc + n;
        float v = acc[s][t][r];
        if (bias) v += bias[n];
        if (Cacc) v += Cacc[ci];
        C[ci] = (CT)v;
      }
    }
  }
}

// ---- embedding gather ------------------------------------------------------
__global__ void k_gather(const int* __restrict__ ids,
                         const float* __restrict__ tok_mu,
                         const float* __restrict__ tok_lv,
                         const float* __restrict__ tok_ra,
                         const float* __restrict__ tok_f,
                         const float* __restrict__ pos_mu,
                         float* mu, float* iv, float* alpha,
                         float* feat, _Float16* feat16)
{
  int row = blockIdx.x;            // b*L + l
  int l   = row & (Ll - 1);
  int t   = ids[row];
  int tid = threadIdx.x;           // 256
  if (tid < DSd){
    mu[row * DSd + tid] = tok_mu[t * DSd + tid] + pos_mu[l * DSd + tid];
    iv[row * DSd + tid] = __expf(-tok_lv[t * DSd + tid]);
  }
  float f = tok_f[t * DFd + tid];
  feat[row * DFd + tid]   = f;
  feat16[row * DFd + tid] = (_Float16)f;
  if (tid == 0) alpha[row] = sigm(tok_ra[t]);
}

// ---- query = mean_L(mu) ----------------------------------------------------
__global__ void k_query(const float* __restrict__ mu, float* query)
{
  int t = threadIdx.x;             // 128 = B*DS
  int b = t >> 6, d = t & 63;
  float s = 0.f;
  for (int l = 0; l < Ll; ++l) s += mu[((b << 10) + l) * DSd + d];
  query[t] = s * (1.0f / Ll);
}

// ---- blob mixture: d2 -> a -> cumprod scan -> wb, t_res, blob_meaning ------
__global__ void k_blob(const float* __restrict__ query,
                       const float* __restrict__ blob_mu,
                       const float* __restrict__ blob_lv,
                       const float* __restrict__ blob_ra,
                       const float* __restrict__ blob_f,
                       float* tres, float* blobm)
{
  __shared__ float qs[DSd];
  __shared__ float s[NBn];
  __shared__ float wsh[NBn];
  int tid = threadIdx.x;           // 512
  for (int b = 0; b < Bb; ++b){
    if (tid < DSd) qs[tid] = query[b * DSd + tid];
    __syncthreads();
    const float* bm = blob_mu + tid * DSd;
    const float* bl = blob_lv + tid * DSd;
    float d2 = 0.f;
    for (int d = 0; d < DSd; ++d){
      float ivd = __expf(-bl[d]);
      float dq  = qs[d] - bm[d];
      d2 += ivd * dq * dq;
    }
    float a = sigm(blob_ra[tid]) * __expf(-0.5f * d2);
    s[tid] = 1.0f - a;
    __syncthreads();
    for (int off = 1; off < NBn; off <<= 1){
      float v = s[tid];
      if (tid >= off) v *= s[tid - off];
      __syncthreads(); s[tid] = v; __syncthreads();
    }
    float excl = tid ? s[tid - 1] : 1.0f;
    wsh[tid] = a * excl;
    if (tid == NBn - 1) tres[b] = s[NBn - 1];
    __syncthreads();
    if (tid < DFd){
      float acc = 0.f;
      for (int n = 0; n < NBn; ++n) acc += wsh[n] * blob_f[n * DFd + tid];
      blobm[b * DFd + tid] = acc;
    }
    __syncthreads();
  }
}

// ---- attention W matrix: per row i, causal transmittance scan --------------
__global__ void k_wmat(const float* __restrict__ q,   // (NROW, H*DS)
                       const float* __restrict__ mu,  // (NROW, DS)
                       const float* __restrict__ iv,
                       const float* __restrict__ alpha,
                       _Float16* __restrict__ W)      // (B*H, L, L)
{
  int i   = blockIdx.x;            // row
  int z   = blockIdx.y;            // b*H + h
  int b   = z >> 3, h = z & 7;
  int tid = threadIdx.x;           // 128
  __shared__ float qs[DSd];
  __shared__ float s[128];
  __shared__ float Tc;
  if (tid < DSd) qs[tid] = q[((long)(b * Ll + i)) * (Hh * DSd) + h * DSd + tid];
  if (tid == 0) Tc = 1.0f;
  __syncthreads();
  _Float16* Wrow = W + ((long)z * Ll + i) * Ll;

  for (int c = 0; c < Ll / 128; ++c){
    int j = c * 128 + tid;
    float a = 0.f;
    if (j <= i){
      const float* mr = mu + ((long)b * Ll + j) * DSd;
      const float* vr = iv + ((long)b * Ll + j) * DSd;
      float d2 = 0.f;
#pragma unroll 8
      for (int d = 0; d < DSd; ++d){
        float dq = qs[d] - mr[d];
        d2 += vr[d] * dq * dq;
      }
      a = alpha[b * Ll + j] * __expf(-0.5f * d2);
    }
    s[tid] = 1.0f - a;
    __syncthreads();
    for (int off = 1; off < 128; off <<= 1){
      float v = s[tid];
      if (tid >= off) v *= s[tid - off];
      __syncthreads(); s[tid] = v; __syncthreads();
    }
    float excl = tid ? s[tid - 1] : 1.0f;
    Wrow[j] = (_Float16)(a * Tc * excl);
    __syncthreads();
    if (tid == 0) Tc = Tc * s[127];
    __syncthreads();
  }
}

// ---- fp32 -> f16, 4-wide (n % 4 == 0) --------------------------------------
__global__ void k_cvt4(const float* __restrict__ src, _Float16* __restrict__ dst, int n4){
  int i = blockIdx.x * 256 + threadIdx.x;
  if (i < n4){
    v4f f = ((const v4f*)src)[i];
    v4h h;
#pragma unroll
    for (int j = 0; j < 4; ++j) h[j] = (_Float16)f[j];
    ((v4h*)dst)[i] = h;
  }
}

// ---- ctx = [feat16 | meaning] f16 ------------------------------------------
__global__ void k_ctx(const _Float16* __restrict__ feat16,
                      const float* __restrict__ meaning, _Float16* __restrict__ ctx){
  int idx = blockIdx.x * 256 + threadIdx.x;      // 2048*512
  int row = idx >> 9, c = idx & 511;
  ctx[idx] = (c < DFd) ? feat16[row * DFd + c]
                       : (_Float16)meaning[row * DFd + (c - DFd)];
}

// ---- alpha gate (N=1 gemv) -------------------------------------------------
__global__ void k_agate(const _Float16* __restrict__ ctx, const float* __restrict__ w,
                        const float* __restrict__ b, float* alpha){
  int row = blockIdx.x; int tid = threadIdx.x;   // 128
  __shared__ float s[128];
  float acc = 0.f;
  for (int c = tid; c < 2 * DFd; c += 128)
    acc += (float)ctx[(long)row * (2 * DFd) + c] * w[c];
  s[tid] = acc; __syncthreads();
  for (int off = 64; off; off >>= 1){ if (tid < off) s[tid] += s[tid + off]; __syncthreads(); }
  if (tid == 0) alpha[row] *= sigm(s[0] + b[0]);
}

// ---- fuse blob + meaning, gate, layernorm -> h (f16) -----------------------
__global__ void k_fuse(const float* __restrict__ meaning, const float* __restrict__ blobm,
                       const float* __restrict__ tres, const float* __restrict__ bg_w,
                       const float* __restrict__ bg_b, const float* __restrict__ ln_g,
                       const float* __restrict__ ln_b, _Float16* __restrict__ hln){
  int row = blockIdx.x; int b = row >> 10; int f = threadIdx.x;  // 256
  __shared__ float s[DFd];
  float bm = blobm[b * DFd + f];
  float mn = meaning[(long)row * DFd + f];
  s[f] = bg_w[f] * bm + bg_w[DFd + f] * mn;
  __syncthreads();
  for (int off = 128; off; off >>= 1){ if (f < off) s[f] += s[f + off]; __syncthreads(); }
  float gate = sigm(s[0] + bg_b[0]);
  __syncthreads();
  float t  = tres[b];
  float fu = (1.0f - t) * gate * bm + t * mn;
  s[f] = fu; __syncthreads();
  for (int off = 128; off; off >>= 1){ if (f < off) s[f] += s[f + off]; __syncthreads(); }
  float mean = s[0] * (1.0f / DFd);
  __syncthreads();
  float d = fu - mean;
  s[f] = d * d; __syncthreads();
  for (int off = 128; off; off >>= 1){ if (f < off) s[f] += s[f + off]; __syncthreads(); }
  float var = s[0] * (1.0f / DFd);
  float h = d * rsqrtf(var + 1e-5f) * ln_g[f] + ln_b[f];
  hln[(long)row * DFd + f] = (_Float16)h;
}

// ============================================================================
extern "C" void kernel_launch(void* const* d_in, const int* in_sizes, int n_in,
                              void* d_out, int out_size, void* d_ws, size_t ws_size,
                              hipStream_t stream)
{
  (void)in_sizes; (void)n_in; (void)out_size; (void)ws_size;
  const int*   token_ids = (const int*)  d_in[0];
  const float* tok_mu    = (const float*)d_in[1];
  const float* tok_lv    = (const float*)d_in[2];
  const float* tok_ra    = (const float*)d_in[3];
  const float* tok_f     = (const float*)d_in[4];
  const float* pos_mu    = (const float*)d_in[5];
  const float* qproj_w   = (const float*)d_in[6];
  const float* qproj_b   = (const float*)d_in[7];
  const float* hproj_w   = (const float*)d_in[8];
  const float* hproj_b   = (const float*)d_in[9];
  const float* mu_up_w   = (const float*)d_in[10];
  const float* mu_up_b   = (const float*)d_in[11];
  const float* agate_w   = (const float*)d_in[12];
  const float* agate_b   = (const float*)d_in[13];
  const float* ffn_w     = (const float*)d_in[14];
  const float* ffn_b     = (const float*)d_in[15];
  const float* ln_g      = (const float*)d_in[16];
  const float* ln_b      = (const float*)d_in[17];
  const float* lm_w      = (const float*)d_in[18];
  const float* lm_b      = (const float*)d_in[19];
  const float* blob_mu   = (const float*)d_in[20];
  const float* blob_lv   = (const float*)d_in[21];
  const float* blob_ra   = (const float*)d_in[22];
  const float* blob_f    = (const float*)d_in[23];
  const float* bgate_w   = (const float*)d_in[24];
  const float* bgate_b   = (const float*)d_in[25];
  float* out = (float*)d_out;

  // workspace carve-up (256B aligned)
  char* base = (char*)d_ws; size_t off = 0;
  auto alloc = [&](size_t bytes)->void*{
    void* p = base + off; off += (bytes + 255) & ~(size_t)255; return p;
  };
  float*    mu      = (float*)   alloc((size_t)NROW * DSd * 4);
  float*    iv      = (float*)   alloc((size_t)NROW * DSd * 4);
  float*    alpha   = (float*)   alloc((size_t)NROW * 4);
  float*    feat    = (float*)   alloc((size_t)NROW * DFd * 4);
  _Float16* feat16  = (_Float16*)alloc((size_t)NROW * DFd * 2);
  _Float16* mu16    = (_Float16*)alloc((size_t)NROW * DSd * 2);
  float*    query   = (float*)   alloc((size_t)Bb * DSd * 4);
  float*    tres    = (float*)   alloc((size_t)Bb * 4);
  float*    blobm   = (float*)   alloc((size_t)Bb * DFd * 4);
  float*    q       = (float*)   alloc((size_t)NROW * Hh * DSd * 4);
  _Float16* W16     = (_Float16*)alloc((size_t)Bb * Hh * Ll * Ll * 2);
  _Float16* mcat16  = (_Float16*)alloc((size_t)NROW * Hh * DFd * 2);
  float*    meaning = (float*)   alloc((size_t)NROW * DFd * 4);
  _Float16* ctx16   = (_Float16*)alloc((size_t)NROW * 2 * DFd * 2);
  _Float16* hln16   = (_Float16*)alloc((size_t)NROW * DFd * 2);
  _Float16* qpw16   = (_Float16*)alloc((size_t)Hh * DSd * DSd * 2);
  _Float16* hpw16   = (_Float16*)alloc((size_t)DFd * Hh * DFd * 2);
  _Float16* muw16   = (_Float16*)alloc((size_t)DSd * 2 * DFd * 2);
  _Float16* ffw16   = (_Float16*)alloc((size_t)DFd * 2 * DFd * 2);
  _Float16* lmw16   = (_Float16*)alloc((size_t)Vv * DFd * 2);

  // 1) gather + small reductions
  k_gather<<<NROW, 256, 0, stream>>>(token_ids, tok_mu, tok_lv, tok_ra, tok_f,
                                     pos_mu, mu, iv, alpha, feat, feat16);
  k_query<<<1, 128, 0, stream>>>(mu, query);
  k_blob <<<1, NBn, 0, stream>>>(query, blob_mu, blob_lv, blob_ra, blob_f, tres, blobm);

  // 2) fp32 -> f16 conversions (all sizes % 4 == 0)
  auto cvt = [&](const float* s, _Float16* d, int n){
    int n4 = n >> 2;
    k_cvt4<<<(n4 + 255) / 256, 256, 0, stream>>>(s, d, n4);
  };
  cvt(qproj_w, qpw16, Hh * DSd * DSd);
  cvt(hproj_w, hpw16, DFd * Hh * DFd);
  cvt(mu_up_w, muw16, DSd * 2 * DFd);
  cvt(ffn_w,   ffw16, DFd * 2 * DFd);
  cvt(lm_w,    lmw16, Vv * DFd);

  // 3) passes
  for (int p = 0; p < 2; ++p){
    cvt(mu, mu16, NROW * DSd);
    // q = mu @ qproj_w^T + b  : M=2048 N=512 K=64
    gemm_tn<float><<<dim3(512/64, NROW/256, 1), 256, 0, stream>>>(
        mu16, qpw16, qproj_b, nullptr, q, NROW, Hh*DSd, DSd, Hh*DSd,
        0, 0, 1, 0, 0, 1);
    // W (B*H, L, L) f16
    k_wmat<<<dim3(Ll, Bb*Hh, 1), 128, 0, stream>>>(q, mu, iv, alpha, W16);
    // m_cat[b,i,h,f] = W @ feat : per z=b*H+h, M=1024 N=256 K=1024, store f16
    gemm_tn<_Float16><<<dim3(DFd/64, Ll/256, Bb*Hh), 256, 0, stream>>>(
        W16, feat16, nullptr, nullptr, mcat16, Ll, DFd, Ll, Hh*DFd,
        (long)Ll*Ll, (long)Ll*DFd, Hh,
        (long)Ll*Hh*DFd, (long)DFd, Hh);
    // meaning = m_cat @ hproj_w^T + b : M=2048 N=256 K=2048
    gemm_tn<float><<<dim3(DFd/64, NROW/256, 1), 256, 0, stream>>>(
        mcat16, hpw16, hproj_b, nullptr, meaning, NROW, DFd, Hh*DFd, DFd,
        0, 0, 1, 0, 0, 1);
    if (p == 0){
      k_ctx<<<(NROW*2*DFd)/256, 256, 0, stream>>>(feat16, meaning, ctx16);
      k_agate<<<NROW, 128, 0, stream>>>(ctx16, agate_w, agate_b, alpha);
      // mu += ctx @ mu_up_w^T + b : N=64 K=512 (in-place accumulate)
      gemm_tn<float><<<dim3(1, NROW/256, 1), 256, 0, stream>>>(
          ctx16, muw16, mu_up_b, mu, mu, NROW, DSd, 2*DFd, DSd,
          0, 0, 1, 0, 0, 1);
      // feat += ctx @ ffn_w^T + b : N=256 K=512
      gemm_tn<float><<<dim3(DFd/64, NROW/256, 1), 256, 0, stream>>>(
          ctx16, ffw16, ffn_b, feat, feat, NROW, DFd, 2*DFd, DFd,
          0, 0, 1, 0, 0, 1);
      cvt(feat, feat16, NROW * DFd);
    }
  }

  // 4) fuse + layernorm -> h (f16)
  k_fuse<<<NROW, DFd, 0, stream>>>(meaning, blobm, tres, bgate_w, bgate_b,
                                   ln_g, ln_b, hln16);
  // 5) logits = h @ lm_w^T + lm_b : M=2048 N=32000 K=256 -> fp32 out
  gemm_tn<float><<<dim3(Vv/64, NROW/256, 1), 256, 0, stream>>>(
      hln16, lmw16, lm_b, nullptr, out, NROW, Vv, DFd, Vv,
      0, 0, 1, 0, 0, 1);
}
